// ScanAssociativeRNNAttention_48137993454031
// MI455X (gfx1250) — matrix operand order repaired and checked by reading, not compiled, not run on hardware
//
#include <hip/hip_runtime.h>
#include <cstdint>

#define BQ 4
#define TQ 4096
#define IQ 1024
#define HQ 16
#define DQ 64
#define NQ 2048   // H*D*2
#define NCH 64    // chunks per sequence
#define CH  64    // timesteps per chunk

typedef __attribute__((ext_vector_type(16))) __bf16 bf16x16;
typedef __attribute__((ext_vector_type(8)))  float  f32x8;
typedef __attribute__((ext_vector_type(4)))  unsigned int u32x4;
typedef __attribute__((ext_vector_type(4)))  int      i32x4;
typedef __attribute__((ext_vector_type(8)))  int      i32x8;

__device__ __forceinline__ unsigned short f2bf(float f) {
  union { float f; unsigned u; } x; x.f = f;
  unsigned r = x.u + 0x7FFFu + ((x.u >> 16) & 1u);   // RNE
  return (unsigned short)(r >> 16);
}

// ---------------------------------------------------------------------------
// Kernel 0a: X fp32 -> bf16 (row-major, 16M elements)
// ---------------------------------------------------------------------------
__global__ void convert_x(const float* __restrict__ X,
                          unsigned short* __restrict__ Xbf)
{
  const size_t i = ((size_t)blockIdx.x * blockDim.x + threadIdx.x) * 8;
  float4 a = *(const float4*)(X + i);
  float4 b = *(const float4*)(X + i + 4);
  unsigned short t[8] = { f2bf(a.x), f2bf(a.y), f2bf(a.z), f2bf(a.w),
                          f2bf(b.x), f2bf(b.y), f2bf(b.z), f2bf(b.w) };
  *(uint4*)(Xbf + i) = *(const uint4*)t;
}

// ---------------------------------------------------------------------------
// Kernel 0b: Wk (I x 2048) fp32 -> bf16 TRANSPOSED WkT (2048 x I)
// so GEMM B tiles become contiguous 64B rows (TDM/WMMA friendly).
// ---------------------------------------------------------------------------
__global__ void convert_wkT(const float* __restrict__ Wk,
                            unsigned short* __restrict__ WkT)
{
  const int idx = blockIdx.x * blockDim.x + threadIdx.x;  // 262144 threads
  const int n  = idx & (NQ - 1);
  const int k8 = idx >> 11;                               // 0..127
  unsigned short t[8];
  #pragma unroll
  for (int j = 0; j < 8; ++j)
    t[j] = f2bf(Wk[(size_t)(k8 * 8 + j) * NQ + n]);
  *(uint4*)(WkT + (size_t)n * IQ + k8 * 8) = *(const uint4*)t;
}

// ---------------------------------------------------------------------------
// TDM 2D tile load: tile (tile_d0=32 x tile_d1=128) of bf16, row stride 1024.
// LDS receives tile_d1 rows x 64B contiguous = [row][32] bf16 layout.
// ---------------------------------------------------------------------------
__device__ __forceinline__ void tdm_load_tile(unsigned lds_off,
                                              const unsigned short* gptr,
                                              int tensor_d1)
{
  const unsigned long long ga = (unsigned long long)(uintptr_t)gptr;
  u32x4 g0;
  g0[0] = 1u;                                          // count=1, user desc
  g0[1] = lds_off;                                     // lds_addr
  g0[2] = (unsigned)(ga & 0xFFFFFFFFu);                // global_addr lo
  g0[3] = (unsigned)((ga >> 32) & 0x1FFFFFFu) | (2u << 30);  // addr hi | type=2
  i32x8 g1;
  g1[0] = 0x00010000;                    // wg_mask=0, data_size=1 (2B)
  g1[1] = (IQ & 0xFFFF) << 16;           // tensor_dim0 lo16 (=1024)
  g1[2] = ((IQ >> 16) & 0xFFFF) | ((tensor_d1 & 0xFFFF) << 16);
  g1[3] = ((tensor_d1 >> 16) & 0xFFFF) | (32 << 16);   // tile_dim0 = 32
  g1[4] = 128;                           // tile_dim1 = 128, tile_dim2 = 0
  g1[5] = IQ;                            // tensor_dim0_stride = 1024 elems
  g1[6] = 0;
  g1[7] = 0;
  i32x4 z4; z4[0] = 0; z4[1] = 0; z4[2] = 0; z4[3] = 0;
#if __clang_major__ >= 23
  i32x8 z8; z8[0]=0; z8[1]=0; z8[2]=0; z8[3]=0; z8[4]=0; z8[5]=0; z8[6]=0; z8[7]=0;
  __builtin_amdgcn_tensor_load_to_lds(g0, g1, z4, z4, z8, 0);
#else
  __builtin_amdgcn_tensor_load_to_lds(g0, g1, z4, z4, 0);
#endif
}

// ---------------------------------------------------------------------------
// Kernel 1: kv = silu(X @ Wk); v -> Wv[(b,h,t,d)], s = q·k via ds_add_f32.
// bf16 WMMA, fp32 accum. Tiles DMA'd by the Tensor Data Mover, double
// buffered, pipelined with s_wait_tensorcnt.
// LDS map (dynamic only, so offsets are literal): A0@0 A1@8192 B0@16384
// B1@24576 (each 8KB), s_acc@32768 (512B). Total 33280B.
// ---------------------------------------------------------------------------
__launch_bounds__(256, 2)
__global__ void kv_gemm_silu(const unsigned short* __restrict__ Xbf,  // (B*T, I) bf16
                             const unsigned short* __restrict__ WkT,  // (2048, I) bf16
                             const float* __restrict__ Q,             // (H, D)
                             float* __restrict__ Wv,                  // (B,H,T,D)
                             float* __restrict__ Ws)                  // (B,H,T)
{
  extern __shared__ __align__(16) unsigned short smem[];
  float* s_acc = (float*)(smem + 16384);   // byte 32768

  const int h     = blockIdx.x;            // head
  const int mtile = blockIdx.y;            // 0..127
  const int row0  = mtile * 128;
  const int b     = row0 >> 12;
  const int t0    = row0 & (TQ - 1);

  const int tid   = threadIdx.x;
  const int lane  = tid & 31;
  const int wave  = tid >> 5;
  const int waveM = wave >> 1;             // 0..3
  const int waveN = wave & 1;              // 0..1

  f32x8 acc[2][4];
  #pragma unroll
  for (int i = 0; i < 2; ++i)
    #pragma unroll
    for (int j = 0; j < 4; ++j)
      acc[i][j] = (f32x8){0.f,0.f,0.f,0.f,0.f,0.f,0.f,0.f};

  const unsigned short* Asrc = Xbf + (size_t)row0 * IQ;       // + k0
  const unsigned short* Bsrc = WkT + (size_t)(h * 128) * IQ;  // + k0

  // prologue: DMA tile 0 into buffer 0
  if (wave == 0)      tdm_load_tile(0u,      Asrc,      BQ * TQ);
  else if (wave == 1) tdm_load_tile(16384u,  Bsrc,      NQ);

  const int l16 = lane & 15;
  const int kcA = (lane >> 4) * 8;     // A frag: K 0-7/16-23 vs 8-15/24-31
  const int kbB = (lane >> 4) * 8;     // B frag: K 0-7/16-23 vs 8-15/24-31

  for (int it = 0; it < IQ / 32; ++it) {
    const int buf = it & 1;
    // DMA next tile into the other buffer (its previous reader finished
    // before the trailing barrier of iteration it-1).
    if (it + 1 < IQ / 32) {
      const int k1 = (it + 1) * 32;
      if (wave == 0)      tdm_load_tile(buf ? 0u : 8192u,       Asrc + k1, BQ * TQ);
      else if (wave == 1) tdm_load_tile(buf ? 16384u : 24576u,  Bsrc + k1, NQ);
      __builtin_amdgcn_s_wait_tensorcnt(1);   // tile `it` complete
    } else {
      __builtin_amdgcn_s_wait_tensorcnt(0);
    }
    __syncthreads();                          // tile `it` visible to all waves

    const unsigned short* As = smem + (buf ? 4096 : 0);
    const unsigned short* Bs = smem + (buf ? 12288 : 8192);

    bf16x16 afrag[2], bfrag[4];
    #pragma unroll
    for (int fm = 0; fm < 2; ++fm) {
      const unsigned short* p = &As[(waveM * 32 + fm * 16 + l16) * 32 + kcA];
      union { uint4 q[2]; bf16x16 v; } u;
      u.q[0] = *(const uint4*)(p);
      u.q[1] = *(const uint4*)(p + 16);
      afrag[fm] = u.v;
    }
    #pragma unroll
    for (int fn = 0; fn < 4; ++fn) {
      const unsigned short* p = &Bs[(waveN * 64 + fn * 16 + l16) * 32 + kbB];
      union { uint4 q[2]; bf16x16 v; } u;
      u.q[0] = *(const uint4*)(p);
      u.q[1] = *(const uint4*)(p + 16);
      bfrag[fn] = u.v;
    }
    #pragma unroll
    for (int fm = 0; fm < 2; ++fm)
      #pragma unroll
      for (int fn = 0; fn < 4; ++fn)
        acc[fm][fn] = __builtin_amdgcn_wmma_f32_16x16x32_bf16(
            false, afrag[fm], false, bfrag[fn], (short)0, acc[fm][fn],
            false, false);
    __syncthreads();                          // done reading buf
  }

  // ---- epilogue: silu, v -> global, s via LDS float atomics ----
  if (tid < 128) s_acc[tid] = 0.0f;
  __syncthreads();

  const int rhalf = (lane >> 4) * 8;          // C rows: VGPR j + 8*(lane>=16)
  #pragma unroll
  for (int fm = 0; fm < 2; ++fm) {
    const int rbase = waveM * 32 + fm * 16 + rhalf;
    #pragma unroll
    for (int fn = 0; fn < 4; ++fn) {
      const int c  = waveN * 64 + fn * 16 + l16;   // col within head (d*2+kk)
      const int d  = c >> 1;
      const int kk = c & 1;
      const float qv = (kk == 0) ? Q[h * DQ + d] : 0.0f;
      #pragma unroll
      for (int j = 0; j < 8; ++j) {
        const int r = rbase + j;
        const float x = acc[fm][fn][j];
        const float y = x * __builtin_amdgcn_rcpf(1.0f + __expf(-x)); // silu
        if (kk == 0) {
          atomicAdd(&s_acc[r], y * qv);            // ds_add_f32
        } else {
          const int t = t0 + r;
          Wv[((size_t)(b * HQ + h) * TQ + t) * DQ + d] = y;
        }
      }
    }
  }
  __syncthreads();
  if (tid < 128)
    Ws[(size_t)(b * HQ + h) * TQ + t0 + tid] = s_acc[tid];
}

// ---------------------------------------------------------------------------
// Kernel 2: per-chunk aggregates (m,u,w[64]). One wave per chunk.
// ---------------------------------------------------------------------------
__global__ void chunk_aggregate(const float* __restrict__ Wv,
                                const float* __restrict__ Ws,
                                float* __restrict__ Am,
                                float* __restrict__ Au,
                                float* __restrict__ Aw)
{
  const int gw = (blockIdx.x * blockDim.x + threadIdx.x) >> 5;
  if (gw >= BQ * HQ * NCH) return;
  const int lane = threadIdx.x & 31;
  const int ch = gw & (NCH - 1);
  const int bh = gw >> 6;
  const float* vbase = Wv + ((size_t)bh * TQ + ch * CH) * DQ;
  const float* sbase = Ws + (size_t)bh * TQ + ch * CH;

  float m = -3.0e38f, u = 0.0f, w0 = 0.0f, w1 = 0.0f;
  for (int t = 0; t < CH; ++t) {
    const float s  = sbase[t];
    const float v0 = vbase[(size_t)t * DQ + lane];
    const float v1 = vbase[(size_t)t * DQ + lane + 32];
    const float mn = fmaxf(m, s);
    const float ea = __expf(m - mn);
    const float eb = __expf(s - mn);
    u  = u  * ea + eb;
    w0 = w0 * ea + v0 * eb;
    w1 = w1 * ea + v1 * eb;
    m  = mn;
  }
  float* awp = Aw + (size_t)gw * DQ;
  awp[lane]      = w0;
  awp[lane + 32] = w1;
  if (lane == 0) { Am[gw] = m; Au[gw] = u; }
}

// ---------------------------------------------------------------------------
// Kernel 3: serial scan of chunk aggregates per (b,h) -> EXCLUSIVE prefixes.
// ---------------------------------------------------------------------------
__global__ void chunk_scan(const float* __restrict__ Am,
                           const float* __restrict__ Au,
                           const float* __restrict__ Aw,
                           float* __restrict__ Pm,
                           float* __restrict__ Pu,
                           float* __restrict__ Pw)
{
  const int bh   = blockIdx.x;
  const int lane = threadIdx.x & 31;
  float m = -3.0e38f, u = 0.0f, w0 = 0.0f, w1 = 0.0f;
  for (int ch = 0; ch < NCH; ++ch) {
    const int idx = bh * NCH + ch;
    float* pwp = Pw + (size_t)idx * DQ;
    pwp[lane]      = w0;
    pwp[lane + 32] = w1;
    if (lane == 0) { Pm[idx] = m; Pu[idx] = u; }

    const float ma = Am[idx];
    const float ua = Au[idx];
    const float* awp = Aw + (size_t)idx * DQ;
    const float a0 = awp[lane];
    const float a1 = awp[lane + 32];
    const float mn = fmaxf(m, ma);
    const float ea = __expf(m  - mn);
    const float eb = __expf(ma - mn);
    u  = u  * ea + ua * eb;
    w0 = w0 * ea + a0 * eb;
    w1 = w1 * ea + a1 * eb;
    m  = mn;
  }
}

// ---------------------------------------------------------------------------
// Kernel 4: re-walk chunk with exclusive prefix; out[(b,t,h,d)] = w/u.
// ---------------------------------------------------------------------------
__global__ void chunk_apply(const float* __restrict__ Wv,
                            const float* __restrict__ Ws,
                            const float* __restrict__ Pm,
                            const float* __restrict__ Pu,
                            const float* __restrict__ Pw,
                            float* __restrict__ out)
{
  const int gw = (blockIdx.x * blockDim.x + threadIdx.x) >> 5;
  if (gw >= BQ * HQ * NCH) return;
  const int lane = threadIdx.x & 31;
  const int ch = gw & (NCH - 1);
  const int bh = gw >> 6;
  const int b = bh >> 4, h = bh & 15;
  const float* vbase = Wv + ((size_t)bh * TQ + ch * CH) * DQ;
  const float* sbase = Ws + (size_t)bh * TQ + ch * CH;
  const float* pwp   = Pw + (size_t)gw * DQ;

  float m = Pm[gw], u = Pu[gw];
  float w0 = pwp[lane], w1 = pwp[lane + 32];
  for (int t = 0; t < CH; ++t) {
    const float s  = sbase[t];
    const float v0 = vbase[(size_t)t * DQ + lane];
    const float v1 = vbase[(size_t)t * DQ + lane + 32];
    const float mn = fmaxf(m, s);
    const float ea = __expf(m - mn);
    const float eb = __expf(s - mn);
    u  = u  * ea + eb;
    w0 = w0 * ea + v0 * eb;
    w1 = w1 * ea + v1 * eb;
    m  = mn;
    const int tg = ch * CH + t;
    float* op = out + (((size_t)b * TQ + tg) * HQ + h) * DQ;
    const float inv = __builtin_amdgcn_rcpf(u);
    op[lane]      = w0 * inv;
    op[lane + 32] = w1 * inv;
  }
}

// ---------------------------------------------------------------------------
extern "C" void kernel_launch(void* const* d_in, const int* in_sizes, int n_in,
                              void* d_out, int out_size, void* d_ws, size_t ws_size,
                              hipStream_t stream) {
  (void)in_sizes; (void)n_in; (void)out_size; (void)ws_size;
  const float* X  = (const float*)d_in[0];  // (B,T,I)
  const float* Wk = (const float*)d_in[1];  // (I,H,D,2)
  const float* Q  = (const float*)d_in[2];  // (H,D)
  float* out = (float*)d_out;

  float* ws = (float*)d_ws;
  float* Wv = ws;                                   // B*H*T*D = 16,777,216 f
  float* Ws = Wv + (size_t)BQ * HQ * TQ * DQ;       // B*H*T   =    262,144 f
  float* Am = Ws + (size_t)BQ * HQ * TQ;            // 4096
  float* Au = Am + BQ * HQ * NCH;                   // 4096
  float* Aw = Au + BQ * HQ * NCH;                   // 4096*64
  float* Pm = Aw + (size_t)BQ * HQ * NCH * DQ;      // 4096
  float* Pu = Pm + BQ * HQ * NCH;                   // 4096
  float* Pw = Pu + BQ * HQ * NCH;                   // 4096*64
  unsigned short* Xbf = (unsigned short*)(Pw + (size_t)BQ * HQ * NCH * DQ);
  unsigned short* WkT = Xbf + (size_t)BQ * TQ * IQ; // 2048 x 1024 bf16

  convert_x  <<<(BQ * TQ * IQ) / 8 / 256, 256, 0, stream>>>(X, Xbf);
  convert_wkT<<<(IQ * NQ) / 8 / 256,      256, 0, stream>>>(Wk, WkT);

  dim3 g1(HQ, (BQ * TQ) / 128);                     // 16 x 128 tiles
  kv_gemm_silu<<<g1, 256, 33280, stream>>>(Xbf, WkT, Q, Wv, Ws);

  const int nchunks = BQ * HQ * NCH;                // 4096 waves
  chunk_aggregate<<<nchunks * 32 / 256, 256, 0, stream>>>(Wv, Ws, Am, Au, Aw);
  chunk_scan<<<BQ * HQ, 32, 0, stream>>>(Am, Au, Aw, Pm, Pu, Pw);
  chunk_apply<<<nchunks * 32 / 256, 256, 0, stream>>>(Wv, Ws, Pm, Pu, Pw, out);
}
